// AAE_GCN_67353677136179
// MI455X (gfx1250) — compile-verified, matrix operand order checked by reading, not run on hardware
//
#include <hip/hip_runtime.h>
#include <hip/hip_bf16.h>

// ---------------------------------------------------------------------------
// Types / helpers
// ---------------------------------------------------------------------------
typedef __attribute__((ext_vector_type(16))) __bf16 v16bf;
typedef __attribute__((ext_vector_type(8)))  float  v8f;

union Frag16 { unsigned int u32[8]; unsigned short us[16]; v16bf v; };

__device__ __forceinline__ unsigned short f2bf(float f) {
    unsigned int u = __float_as_uint(f);
    u += 0x7FFFu + ((u >> 16) & 1u);           // round-to-nearest-even
    return (unsigned short)(u >> 16);
}
__device__ __forceinline__ float bf2f(unsigned short h) {
    return __uint_as_float(((unsigned int)h) << 16);
}
__device__ __forceinline__ float lrelu(float v) { return v > 0.f ? v : 0.01f * v; }

// ---------------------------------------------------------------------------
// GCN kernels
// ---------------------------------------------------------------------------
__global__ void k_init_deg(float* deg, int N) {
    int n = blockIdx.x * blockDim.x + threadIdx.x;
    if (n < N) deg[n] = 2.0f;                  // improved=True self-loop weight
}

__global__ void k_deg_edges(const int* dst, float* deg, int E) {
    int e = blockIdx.x * blockDim.x + threadIdx.x;
    if (e < E) atomicAdd(&deg[dst[e]], 1.0f);
}

__global__ void k_dinv(float* deg, int N) {
    int n = blockIdx.x * blockDim.x + threadIdx.x;
    if (n < N) {
        float d = deg[n];
        deg[n] = d > 0.f ? __frsqrt_rn(d) : 0.f;
    }
}

// data,noise: [B,N] row-major.  Build xlT[N,B] = (data+0.1*noise)*w_gcn and
// seed aggT[N,B] with the self-loop contribution xl * (2*dinv[n]^2).
__global__ void k_build_xlt(const float* __restrict__ data,
                            const float* __restrict__ noise,
                            const float* __restrict__ wg_p,
                            const float* __restrict__ dinv,
                            float* __restrict__ xlT,
                            float* __restrict__ aggT, int N, int B) {
    __shared__ float tile[32][33];
    const float wg = wg_p[0];
    int tx = threadIdx.x, ty = threadIdx.y;            // block (32,8)
    int n0 = blockIdx.x * 32, b0 = blockIdx.y * 32;
    #pragma unroll
    for (int i = 0; i < 4; ++i) {
        int b = b0 + ty + i * 8, n = n0 + tx;
        size_t o = (size_t)b * N + n;
        tile[ty + i * 8][tx] = (data[o] + 0.1f * noise[o]) * wg;   // tile[b][n]
    }
    __syncthreads();
    #pragma unroll
    for (int i = 0; i < 4; ++i) {
        int n = n0 + ty + i * 8, b = b0 + tx;
        float v  = tile[tx][ty + i * 8];
        float di = dinv[n];
        size_t o = (size_t)n * B + b;
        xlT[o]  = v;
        aggT[o] = v * (2.0f * di * di);
    }
}

// One block per edge, lanes over batch: aggT[:,dst] += xlT[:,src]*dinv[s]*dinv[d]
__global__ void k_scatter(const int* __restrict__ src, const int* __restrict__ dst,
                          const float* __restrict__ dinv,
                          const float* __restrict__ xlT,
                          float* __restrict__ aggT, int B) {
    int e = blockIdx.x;
    int s = src[e], d = dst[e];
    float c = dinv[s] * dinv[d];
    const float* xs = xlT + (size_t)s * B;
    float* ad = aggT + (size_t)d * B;
    for (int b = threadIdx.x; b < B; b += blockDim.x)
        atomicAdd(&ad[b], xs[b] * c);
}

// A_bf16[B,N] = bf16(lrelu(aggT[N,B]^T + b_gcn))
__global__ void k_finalize_gcn(const float* __restrict__ aggT,
                               const float* __restrict__ bg_p,
                               unsigned short* __restrict__ Abf, int N, int B) {
    __shared__ float tile[32][33];
    const float bg = bg_p[0];
    int tx = threadIdx.x, ty = threadIdx.y;            // block (32,8)
    int n0 = blockIdx.x * 32, b0 = blockIdx.y * 32;
    #pragma unroll
    for (int i = 0; i < 4; ++i) {
        int n = n0 + ty + i * 8, b = b0 + tx;
        tile[ty + i * 8][tx] = aggT[(size_t)n * B + b];            // tile[n][b]
    }
    __syncthreads();
    #pragma unroll
    for (int i = 0; i < 4; ++i) {
        int b = b0 + ty + i * 8, n = n0 + tx;
        Abf[(size_t)b * N + n] = f2bf(lrelu(tile[tx][ty + i * 8] + bg));
    }
}

// ---------------------------------------------------------------------------
// f32 [K,N] -> bf16 transposed [N,K] (LDS-tiled) so GEMM B-fragments are
// contiguous in K per lane (vectorizable into global_load_b128).
// ---------------------------------------------------------------------------
__global__ void k_cvt_bf16_t(const float* __restrict__ in,
                             unsigned short* __restrict__ out, int K, int N) {
    __shared__ float tile[32][33];
    int tx = threadIdx.x, ty = threadIdx.y;            // block (32,8)
    int n0 = blockIdx.x * 32, k0 = blockIdx.y * 32;
    #pragma unroll
    for (int i = 0; i < 4; ++i) {
        int k = k0 + ty + i * 8, n = n0 + tx;
        tile[ty + i * 8][tx] = in[(size_t)k * N + n];              // tile[k][n]
    }
    __syncthreads();
    #pragma unroll
    for (int i = 0; i < 4; ++i) {
        int n = n0 + ty + i * 8, k = k0 + tx;
        out[(size_t)n * K + k] = f2bf(tile[tx][ty + i * 8]);
    }
}

// ---------------------------------------------------------------------------
// WMMA bf16 GEMM: C = lrelu(A[MxK] @ W + bias), W given as Bt[N,K] bf16
// (transposed, row-major).  Each wave computes a 32x32 C tile:
// 2 M-subtiles x 2 N-subtiles, 4 accumulators, 4 WMMAs per 32-wide K step.
// One fragment set per iteration; the compiler unrolls x2 and pipelines the
// next step's loads between the WMMAs (verified round 2; no spills).
// 16-bit fragment layout (per lane, half = lane>>4, idx = lane&15):
//   vgpr v holds K pair at k = (v>=4?16:0) + half*8 + (v&3)*2
//   -> dword offsets {0..3, 8..11}; half folded into the per-lane base so
//      each fragment is exactly two global_load_b128 (offsets 0 and 32).
// C layout: vgpr d -> row d (lanes 0-15) / d+8 (lanes 16-31), col = idx.
// ---------------------------------------------------------------------------
__global__ __launch_bounds__(256, 2)
void k_gemm_wmma(const unsigned short* __restrict__ A,
                 const unsigned short* __restrict__ Bt,
                 const float* __restrict__ bias,
                 float* __restrict__ Cf,
                 unsigned short* __restrict__ Cb,
                 int M, int K, int N) {
    const int lane = threadIdx.x & 31;
    const int wave = threadIdx.x >> 5;
    const int tilesN = N >> 5;
    const int tilesTotal = (M >> 5) * tilesN;
    const int tile = blockIdx.x * 8 + wave;            // wave-uniform
    if (tile >= tilesTotal) return;

    const int tm = tile / tilesN, tn = tile % tilesN;
    const int half = lane >> 4, idx = lane & 15;
    const int rowA0 = tm * 32 + idx, rowA1 = rowA0 + 16;
    const int colB0 = tn * 32 + idx, colB1 = colB0 + 16;

    v8f acc00, acc01, acc10, acc11;
    const float bv0 = bias ? bias[colB0] : 0.0f;
    const float bv1 = bias ? bias[colB1] : 0.0f;
    #pragma unroll
    for (int d = 0; d < 8; ++d) {
        acc00[d] = bv0; acc10[d] = bv0;
        acc01[d] = bv1; acc11[d] = bv1;
    }

    // Per-lane bases with the half-offset folded in.
    const unsigned int* pA0 = (const unsigned int*)(A  + (size_t)rowA0 * K) + half * 4;
    const unsigned int* pA1 = (const unsigned int*)(A  + (size_t)rowA1 * K) + half * 4;
    const unsigned int* pB0 = (const unsigned int*)(Bt + (size_t)colB0 * K) + half * 4;
    const unsigned int* pB1 = (const unsigned int*)(Bt + (size_t)colB1 * K) + half * 4;

    #define LOAD_FRAG(f, p)                                         \
        do {                                                        \
            _Pragma("unroll")                                       \
            for (int v = 0; v < 4; ++v) (f).u32[v]     = (p)[v];    \
            _Pragma("unroll")                                       \
            for (int v = 0; v < 4; ++v) (f).u32[4 + v] = (p)[8 + v];\
        } while (0)

    const int steps = K >> 5;
    for (int s = 0; s < steps; ++s) {
        // speculative prefetch two steps ahead (stays inside the workspace)
        __builtin_prefetch((const void*)(pA0 + 32), 0, 3);
        __builtin_prefetch((const void*)(pB0 + 32), 0, 3);
        Frag16 fa0, fa1, fb0, fb1;
        LOAD_FRAG(fa0, pA0); LOAD_FRAG(fa1, pA1);
        LOAD_FRAG(fb0, pB0); LOAD_FRAG(fb1, pB1);
        acc00 = __builtin_amdgcn_wmma_f32_16x16x32_bf16(
            false, fa0.v, false, fb0.v, (short)0, acc00, false, false);
        acc01 = __builtin_amdgcn_wmma_f32_16x16x32_bf16(
            false, fa0.v, false, fb1.v, (short)0, acc01, false, false);
        acc10 = __builtin_amdgcn_wmma_f32_16x16x32_bf16(
            false, fa1.v, false, fb0.v, (short)0, acc10, false, false);
        acc11 = __builtin_amdgcn_wmma_f32_16x16x32_bf16(
            false, fa1.v, false, fb1.v, (short)0, acc11, false, false);
        pA0 += 16; pA1 += 16; pB0 += 16; pB1 += 16;    // next 32-K window
    }
    #undef LOAD_FRAG

    // epilogue: fused leaky-relu, optional f32 / bf16 outputs
    #pragma unroll
    for (int sub = 0; sub < 4; ++sub) {
        const v8f& acc = (sub == 0) ? acc00 : (sub == 1) ? acc01
                        : (sub == 2) ? acc10 : acc11;
        const int rbase = tm * 32 + ((sub >> 1) ? 16 : 0) + half * 8;
        const int col   = (sub & 1) ? colB1 : colB0;
        #pragma unroll
        for (int d = 0; d < 8; ++d) {
            float v = lrelu(acc[d]);
            size_t o = (size_t)(rbase + d) * N + col;
            if (Cf) Cf[o] = v;
            if (Cb) Cb[o] = f2bf(v);
        }
    }
}

// ---------------------------------------------------------------------------
// Discriminator head: d = sigmoid(g2[M,K] @ w[K,1] + b)
// ---------------------------------------------------------------------------
__global__ void k_disc_head(const unsigned short* __restrict__ g2,
                            const float* __restrict__ w,
                            const float* __restrict__ b,
                            float* __restrict__ out, int M, int K) {
    int r = blockIdx.x * blockDim.x + threadIdx.x;
    if (r >= M) return;
    float s = b[0];
    const unsigned short* row = g2 + (size_t)r * K;
    for (int k = 0; k < K; ++k) s += bf2f(row[k]) * w[k];
    out[r] = 1.0f / (1.0f + __expf(-s));
}

// ---------------------------------------------------------------------------
// Host orchestration
// ---------------------------------------------------------------------------
static inline size_t al256(size_t x) { return (x + 255) & ~(size_t)255; }

extern "C" void kernel_launch(void* const* d_in, const int* in_sizes, int n_in,
                              void* d_out, int out_size, void* d_ws, size_t ws_size,
                              hipStream_t stream) {
    (void)n_in; (void)out_size; (void)ws_size;
    const int B = 512;
    const int N = in_sizes[0] / B;          // 20000
    const int E = in_sizes[2] / 2;          // 640000
    const int NE1 = 512, NE2 = 256, NZ = 64, ND2 = 256, ND1 = 512, H = 256;

    const float* data  = (const float*)d_in[0];
    const float* noise = (const float*)d_in[1];
    const int*   eidx  = (const int*)d_in[2];
    const int*   esrc  = eidx;
    const int*   edst  = eidx + E;
    const float* w_gcn = (const float*)d_in[3];
    const float* b_gcn = (const float*)d_in[4];
    const float* enc1_w = (const float*)d_in[5];  const float* enc1_b = (const float*)d_in[6];
    const float* enc2_w = (const float*)d_in[7];  const float* enc2_b = (const float*)d_in[8];
    const float* enc3_w = (const float*)d_in[9];  const float* enc3_b = (const float*)d_in[10];
    const float* dec1_w = (const float*)d_in[11]; const float* dec1_b = (const float*)d_in[12];
    const float* dec2_w = (const float*)d_in[13]; const float* dec2_b = (const float*)d_in[14];
    const float* dec3_w = (const float*)d_in[15]; const float* dec3_b = (const float*)d_in[16];
    const float* dis1_w = (const float*)d_in[17]; const float* dis1_b = (const float*)d_in[18];
    const float* dis2_w = (const float*)d_in[19]; const float* dis2_b = (const float*)d_in[20];
    const float* dis3_w = (const float*)d_in[21]; const float* dis3_b = (const float*)d_in[22];

    float* out      = (float*)d_out;
    float* out_dec3 = out;                                    // [B, N]
    float* out_z    = out + (size_t)B * N;                    // [B, 64]
    float* out_d    = out_z + (size_t)B * NZ;                 // [B]

    // -------- workspace plan (bump, with reuse of the xlT region) ----------
    char* ws = (char*)d_ws;
    const size_t szXLT = (size_t)N * B * 4;                   // 41 MB
    const size_t szABF = (size_t)N * B * 2;                   // 20.5 MB
    size_t off = 0;
    float* xlT  = (float*)(ws + off);           off = al256(off + szXLT);
    float* aggT = (float*)(ws + off);           off = al256(off + szXLT);
    unsigned short* Abf = (unsigned short*)(ws + off); off = al256(off + szABF);
    float* dinv = (float*)(ws + off);           off = al256(off + (size_t)N * 4);
    // small bf16 buffers (transposed weights + activations)
    auto bump = [&](size_t elems) {
        unsigned short* p = (unsigned short*)(ws + off);
        off = al256(off + elems * 2);
        return p;
    };
    unsigned short* enc2_wb = bump((size_t)NE1 * NE2);
    unsigned short* enc3_wb = bump((size_t)NE2 * NZ);
    unsigned short* dec1_wb = bump((size_t)NZ * ND2);
    unsigned short* dec2_wb = bump((size_t)ND2 * ND1);
    unsigned short* dis1_wb = bump((size_t)NZ * H);
    unsigned short* dis2_wb = bump((size_t)H * H);
    unsigned short* h1b = bump((size_t)B * NE1);
    unsigned short* h2b = bump((size_t)B * NE2);
    unsigned short* zb  = bump((size_t)B * NZ);
    unsigned short* d1b = bump((size_t)B * ND2);
    unsigned short* d2b = bump((size_t)B * ND1);
    unsigned short* g1b = bump((size_t)B * H);
    unsigned short* g2b = bump((size_t)B * H);
    bump(64);   // tail guard so speculative reads/prefetch past g2b stay in ws
    // big transposed bf16 weight buffer reuses the xlT region post-GCN
    unsigned short* bigWb = (unsigned short*)ws;              // 20.5 MB needed

    // ---------------------------- GCN --------------------------------------
    k_init_deg <<<(N + 255) / 256, 256, 0, stream>>>(dinv, N);
    k_deg_edges<<<(E + 255) / 256, 256, 0, stream>>>(edst, dinv, E);
    k_dinv     <<<(N + 255) / 256, 256, 0, stream>>>(dinv, N);

    dim3 tb(32, 8), tg(N / 32, B / 32);
    k_build_xlt<<<tg, tb, 0, stream>>>(data, noise, w_gcn, dinv, xlT, aggT, N, B);
    k_scatter  <<<E, 256, 0, stream>>>(esrc, edst, dinv, xlT, aggT, B);
    k_finalize_gcn<<<tg, tb, 0, stream>>>(aggT, b_gcn, Abf, N, B);
    // xlT / aggT regions free from here.

    // ---------------- transposed weight conversion + GEMM helper ------------
    auto cvt_t = [&](const float* src, unsigned short* dst, int K, int Nn) {
        dim3 g(Nn / 32, K / 32);
        k_cvt_bf16_t<<<g, tb, 0, stream>>>(src, dst, K, Nn);
    };
    auto gemm = [&](const unsigned short* A, const unsigned short* Bt,
                    const float* bias, float* Cf, unsigned short* Cb,
                    int M, int K, int Nn) {
        int tiles = (M / 32) * (Nn / 32);
        k_gemm_wmma<<<(tiles + 7) / 8, 256, 0, stream>>>(A, Bt, bias, Cf, Cb, M, K, Nn);
    };

    // ------------------------------ encoder ---------------------------------
    cvt_t(enc1_w, bigWb, N, NE1);
    gemm(Abf, bigWb, enc1_b, nullptr, h1b, B, N, NE1);        // h1 [512,512]
    cvt_t(enc2_w, enc2_wb, NE1, NE2);
    gemm(h1b, enc2_wb, enc2_b, nullptr, h2b, B, NE1, NE2);    // h2 [512,256]
    cvt_t(enc3_w, enc3_wb, NE2, NZ);
    gemm(h2b, enc3_wb, enc3_b, out_z, zb, B, NE2, NZ);        // z  [512,64]

    // ------------------------------ decoder ---------------------------------
    cvt_t(dec1_w, dec1_wb, NZ, ND2);
    gemm(zb, dec1_wb, dec1_b, nullptr, d1b, B, NZ, ND2);      // d1 [512,256]
    cvt_t(dec2_w, dec2_wb, ND2, ND1);
    gemm(d1b, dec2_wb, dec2_b, nullptr, d2b, B, ND2, ND1);    // d2 [512,512]
    cvt_t(dec3_w, bigWb, ND1, N);
    gemm(d2b, bigWb, dec3_b, out_dec3, nullptr, B, ND1, N);   // dec_h3 [512,20000]

    // --------------------------- discriminator ------------------------------
    cvt_t(dis1_w, dis1_wb, NZ, H);
    gemm(zb, dis1_wb, dis1_b, nullptr, g1b, B, NZ, H);        // g1 [512,256]
    cvt_t(dis2_w, dis2_wb, H, H);
    gemm(g1b, dis2_wb, dis2_b, nullptr, g2b, B, H, H);        // g2 [512,256]
    k_disc_head<<<(B + 255) / 256, 256, 0, stream>>>(g2b, dis3_w, dis3_b, out_d, B, H);
}
